// RnnAttentionDecoder_29901562314978
// MI455X (gfx1250) — matrix operand order repaired and checked by reading, not compiled, not run on hardware
//
#include <hip/hip_runtime.h>
#include <hip/hip_bf16.h>
#include <math.h>

typedef __bf16 bf16_t;
typedef __attribute__((ext_vector_type(16))) __bf16 v16bf;
typedef __attribute__((ext_vector_type(8)))  __bf16 v8bf;
typedef __attribute__((ext_vector_type(8)))  float  v8f;

// Problem dims (fixed by the reference)
#define B_   128
#define S_   256
#define I_   512
#define H_   512
#define E_   256
#define T_   32
#define V_   6000
#define OUT_ 6001
#define SOS_ 6001      // V+1
#define XHW_ 1280      // I + E + H : concatenated activation row [context|emb|h]
#define G4H_ 2048      // 4*H

__device__ __forceinline__ bf16_t f2bf(float f) {
  // round-to-nearest-even f32 -> bf16 via integer math
  unsigned u = __float_as_uint(f);
  unsigned r = (u + 0x7FFFu + ((u >> 16) & 1u)) >> 16;
  unsigned short s = (unsigned short)r;
  return __builtin_bit_cast(bf16_t, s);
}

__device__ __forceinline__ float sigf(float x) { return 1.0f / (1.0f + __expf(-x)); }

// Load one 16-lane-worth bf16 WMMA operand fragment (A-layout) from a row base.
__device__ __forceinline__ v16bf load_afrag(const bf16_t* row, int kk, int half) {
  v8bf a0 = *(const v8bf*)(row + kk + half * 8);
  v8bf a1 = *(const v8bf*)(row + kk + 16 + half * 8);
  v16bf a;
  #pragma unroll
  for (int i = 0; i < 8; ++i) { a[i] = a0[i]; a[i + 8] = a1[i]; }
  return a;
}
// B-layout fragment: per-lane K values at half*8 and 16 + half*8 (same order as A).
__device__ __forceinline__ v16bf load_bfrag(const bf16_t* row, int kk, int half) {
  v8bf b0 = *(const v8bf*)(row + kk + half * 8);
  v8bf b1 = *(const v8bf*)(row + kk + 16 + half * 8);
  v16bf b;
  #pragma unroll
  for (int i = 0; i < 8; ++i) { b[i] = b0[i]; b[i + 8] = b1[i]; }
  return b;
}

// ---------------------------------------------------------------------------
// Generic WMMA GEMM:  C[M,N] = A[M,K] @ W[N,K]^T (+bias)
// One wave per 16x16 C tile -- used for the skinny per-step GEMMs (M=128)
// where wave count, not L2 bandwidth, is the scarce resource.
// ---------------------------------------------------------------------------
__global__ __launch_bounds__(128) void wmma_gemm_bf16(
    const bf16_t* __restrict__ A, int lda,
    const bf16_t* __restrict__ W, int ldw,
    const float*  __restrict__ bias,
    float* __restrict__ C, int M, int N, int K)
{
  const int wave = threadIdx.x >> 5;
  const int lane = threadIdx.x & 31;
  const int tilesN = (N + 15) >> 4;
  const int tilesM = M >> 4;                // M always a multiple of 16 here
  const int tile = blockIdx.x * 4 + wave;
  if (tile >= tilesM * tilesN) return;
  const int mt = tile / tilesN;
  const int nt = tile - mt * tilesN;
  const int l16  = lane & 15;
  const int half = lane >> 4;
  const int n = nt * 16 + l16;
  const bool nok = (n < N);                 // tail guard (N=6001)
  const bf16_t* arow = A + (size_t)(mt * 16 + l16) * (size_t)lda;
  const bf16_t* wrow = W + (size_t)(nok ? n : 0) * (size_t)ldw;

  v8f acc = {};
  #pragma unroll 4
  for (int kk = 0; kk < K; kk += 32) {
    v16bf a = load_afrag(arow, kk, half);
    v16bf b = load_bfrag(wrow, kk, half);
    acc = __builtin_amdgcn_wmma_f32_16x16x32_bf16(false, a, false, b, (short)0, acc, false, false);
  }
  if (!nok) return;
  const float bv = bias ? bias[n] : 0.0f;
  #pragma unroll
  for (int r = 0; r < 8; ++r) {             // C/D: M = r + 8*half, N = l16
    C[(size_t)(mt * 16 + r + 8 * half) * (size_t)N + n] = acc[r] + bv;
  }
}

// ---------------------------------------------------------------------------
// Register-blocked WMMA GEMM: one wave computes a 32x32 macro-tile
// (2 A-frags x 2 B-frags -> 4 wmma per K-step; 2x the FLOP/byte).
// Requires M % 32 == 0, N % 32 == 0. Used for the big h_proj GEMM.
// ---------------------------------------------------------------------------
__global__ __launch_bounds__(128) void wmma_gemm_bf16_2x2(
    const bf16_t* __restrict__ A, int lda,
    const bf16_t* __restrict__ W, int ldw,
    const float*  __restrict__ bias,
    float* __restrict__ C, int M, int N, int K)
{
  const int wave = threadIdx.x >> 5;
  const int lane = threadIdx.x & 31;
  const int tilesN = N >> 5;
  const int tilesM = M >> 5;
  const int tile = blockIdx.x * 4 + wave;
  if (tile >= tilesM * tilesN) return;
  const int mt = tile / tilesN;
  const int nt = tile - mt * tilesN;
  const int l16  = lane & 15;
  const int half = lane >> 4;
  const bf16_t* arow0 = A + (size_t)(mt * 32 + l16) * (size_t)lda;
  const bf16_t* arow1 = arow0 + (size_t)16 * (size_t)lda;
  const bf16_t* wrow0 = W + (size_t)(nt * 32 + l16) * (size_t)ldw;
  const bf16_t* wrow1 = wrow0 + (size_t)16 * (size_t)ldw;

  v8f c00 = {}, c01 = {}, c10 = {}, c11 = {};
  #pragma unroll 2
  for (int kk = 0; kk < K; kk += 32) {
    v16bf a0 = load_afrag(arow0, kk, half);
    v16bf a1 = load_afrag(arow1, kk, half);
    v16bf b0 = load_bfrag(wrow0, kk, half);
    v16bf b1 = load_bfrag(wrow1, kk, half);
    c00 = __builtin_amdgcn_wmma_f32_16x16x32_bf16(false, a0, false, b0, (short)0, c00, false, false);
    c01 = __builtin_amdgcn_wmma_f32_16x16x32_bf16(false, a0, false, b1, (short)0, c01, false, false);
    c10 = __builtin_amdgcn_wmma_f32_16x16x32_bf16(false, a1, false, b0, (short)0, c10, false, false);
    c11 = __builtin_amdgcn_wmma_f32_16x16x32_bf16(false, a1, false, b1, (short)0, c11, false, false);
  }
  const int n0 = nt * 32 + l16, n1 = n0 + 16;
  const float bv0 = bias ? bias[n0] : 0.0f;
  const float bv1 = bias ? bias[n1] : 0.0f;
  #pragma unroll
  for (int r = 0; r < 8; ++r) {
    const size_t r0 = (size_t)(mt * 32 + r + 8 * half) * (size_t)N;
    const size_t r1 = (size_t)(mt * 32 + 16 + r + 8 * half) * (size_t)N;
    C[r0 + n0] = c00[r] + bv0;
    C[r0 + n1] = c01[r] + bv1;
    C[r1 + n0] = c10[r] + bv0;
    C[r1 + n1] = c11[r] + bv1;
  }
}

// ---------------------------------------------------------------------------
// Attention: e[b,s] = sum_h tanh(hproj[b,s,h]+q[b,h])*w_score[h];
// softmax over s; context[b,i] = sum_s alpha*ctx[b,s,i] -> bf16 into xh[:,0:512]
// One block per batch element; 256 threads (= S).
// ---------------------------------------------------------------------------
__global__ __launch_bounds__(256) void attn_kernel(
    const float* __restrict__ hproj,   // [B,S,H]
    const float* __restrict__ q,       // [B,H]
    const float* __restrict__ wscore,  // [H]
    const float* __restrict__ ctx,     // [B,S,I]
    bf16_t* __restrict__ xh)           // [B,XHW]
{
  __shared__ float q_s[H_], w_s[H_], e_s[S_], red_s[S_];
  const int b = blockIdx.x, tid = threadIdx.x;
  q_s[tid]        = q[b * H_ + tid];
  q_s[tid + 256]  = q[b * H_ + tid + 256];
  w_s[tid]        = wscore[tid];
  w_s[tid + 256]  = wscore[tid + 256];
  __syncthreads();

  // score for s = tid
  const float* hp = hproj + ((size_t)b * S_ + tid) * H_;
  float e = 0.0f;
  #pragma unroll 4
  for (int h = 0; h < H_; ++h) e += tanhf(hp[h] + q_s[h]) * w_s[h];

  // block softmax over 256 scores
  red_s[tid] = e; __syncthreads();
  for (int off = 128; off; off >>= 1) {
    if (tid < off) red_s[tid] = fmaxf(red_s[tid], red_s[tid + off]);
    __syncthreads();
  }
  const float m = red_s[0]; __syncthreads();
  const float ex = __expf(e - m);
  e_s[tid] = ex;
  red_s[tid] = ex; __syncthreads();
  for (int off = 128; off; off >>= 1) {
    if (tid < off) red_s[tid] += red_s[tid + off];
    __syncthreads();
  }
  const float inv = 1.0f / red_s[0];

  // context: threads stride i (coalesced per s); 2 outputs per thread
  float a0 = 0.0f, a1 = 0.0f;
  const float* cb = ctx + (size_t)b * S_ * I_;
  for (int s = 0; s < S_; ++s) {
    const float al = e_s[s];
    a0 += al * cb[(size_t)s * I_ + tid];
    a1 += al * cb[(size_t)s * I_ + tid + 256];
  }
  xh[(size_t)b * XHW_ + tid]        = f2bf(a0 * inv);
  xh[(size_t)b * XHW_ + tid + 256]  = f2bf(a1 * inv);
}

// embedding gather into xh[:,512:768]
__global__ __launch_bounds__(256) void emb_kernel(
    const int* __restrict__ labels, const bf16_t* __restrict__ emb,
    bf16_t* __restrict__ xh, int t)
{
  const int b = blockIdx.x, e = threadIdx.x;
  const int token = (t == 0) ? SOS_ : labels[(t - 1) * B_ + b];
  xh[(size_t)b * XHW_ + I_ + e] = emb[(size_t)token * E_ + e];
}

// LSTM cell: gate order (i,f,g,o); writes h,c f32 and h bf16 into xh[:,768:1280]
__global__ __launch_bounds__(256) void lstm_kernel(
    const float* __restrict__ gates, float* __restrict__ h, float* __restrict__ c,
    bf16_t* __restrict__ xh)
{
  const int b = blockIdx.x;
  const float* g = gates + (size_t)b * G4H_;
  for (int j = threadIdx.x; j < H_; j += 256) {
    const float ig = sigf(g[j]);
    const float fg = sigf(g[H_ + j]);
    const float gg = tanhf(g[2 * H_ + j]);
    const float og = sigf(g[3 * H_ + j]);
    const float cn = fg * c[b * H_ + j] + ig * gg;
    const float hn = og * tanhf(cn);
    c[b * H_ + j] = cn;
    h[b * H_ + j] = hn;
    xh[(size_t)b * XHW_ + I_ + E_ + j] = f2bf(hn);
  }
}

// row-wise log_softmax over OUT_=6001, one block per batch row
__global__ __launch_bounds__(256) void lsm_kernel(
    const float* __restrict__ logits, float* __restrict__ out)
{
  __shared__ float red[256];
  const int b = blockIdx.x, tid = threadIdx.x;
  const float* row = logits + (size_t)b * OUT_;
  float m = -INFINITY;
  for (int o = tid; o < OUT_; o += 256) m = fmaxf(m, row[o]);
  red[tid] = m; __syncthreads();
  for (int off = 128; off; off >>= 1) {
    if (tid < off) red[tid] = fmaxf(red[tid], red[tid + off]);
    __syncthreads();
  }
  m = red[0]; __syncthreads();
  float s = 0.0f;
  for (int o = tid; o < OUT_; o += 256) s += __expf(row[o] - m);
  red[tid] = s; __syncthreads();
  for (int off = 128; off; off >>= 1) {
    if (tid < off) red[tid] += red[tid + off];
    __syncthreads();
  }
  const float lse = m + __logf(red[0]);
  float* orow = out + (size_t)b * OUT_;
  for (int o = tid; o < OUT_; o += 256) orow[o] = row[o] - lse;
}

// f32 -> bf16 bulk convert
__global__ void cvt_kernel(const float* __restrict__ in, bf16_t* __restrict__ out, long n) {
  long i = (long)blockIdx.x * blockDim.x + threadIdx.x;
  if (i < n) out[i] = f2bf(in[i]);
}

// build w_cat = [w_ih | w_hh] bf16 [2048,1280] and b_cat = b_ih + b_hh
__global__ void pack_cat_kernel(const float* __restrict__ w_ih, const float* __restrict__ w_hh,
                                const float* __restrict__ b_ih, const float* __restrict__ b_hh,
                                bf16_t* __restrict__ wcat, float* __restrict__ bcat) {
  long i = (long)blockIdx.x * blockDim.x + threadIdx.x;
  const long n = (long)G4H_ * XHW_;
  if (i < n) {
    const int r = (int)(i / XHW_), c = (int)(i % XHW_);
    const float v = (c < I_ + E_) ? w_ih[(size_t)r * (I_ + E_) + c]
                                  : w_hh[(size_t)r * H_ + (c - I_ - E_)];
    wcat[i] = f2bf(v);
  }
  if (i < G4H_) bcat[i] = b_ih[i] + b_hh[i];
}

__global__ void zero_kernel(float* __restrict__ p, long n) {
  long i = (long)blockIdx.x * blockDim.x + threadIdx.x;
  if (i < n) p[i] = 0.0f;
}

extern "C" void kernel_launch(void* const* d_in, const int* in_sizes, int n_in,
                              void* d_out, int out_size, void* d_ws, size_t ws_size,
                              hipStream_t stream) {
  const float* ctx     = (const float*)d_in[0];
  const int*   lbl     = (const int*)d_in[1];
  /* d_in[2] = num_steps (always 32) */
  const float* emb     = (const float*)d_in[3];
  const float* w_i2h   = (const float*)d_in[4];
  const float* w_h2h   = (const float*)d_in[5];
  const float* b_h2h   = (const float*)d_in[6];
  const float* w_score = (const float*)d_in[7];
  const float* w_ih    = (const float*)d_in[8];
  const float* b_ih    = (const float*)d_in[9];
  const float* w_hh    = (const float*)d_in[10];
  const float* b_hh    = (const float*)d_in[11];
  const float* w_gen   = (const float*)d_in[12];
  const float* b_gen   = (const float*)d_in[13];
  float* out = (float*)d_out;

  // ---- workspace carving (all 256B aligned; sizes are multiples of 256B) ----
  char* ws = (char*)d_ws;
  size_t ofs = 0;
  auto carve = [&](size_t bytes) -> char* {
    char* p = ws + ofs;
    ofs += (bytes + 255) & ~(size_t)255;
    return p;
  };
  bf16_t* ctx_b  = (bf16_t*)carve((size_t)B_ * S_ * I_ * 2);      // 33.5 MB
  float*  hproj  = (float*) carve((size_t)B_ * S_ * H_ * 4);      // 67 MB (L2-resident)
  bf16_t* wi2h_b = (bf16_t*)carve((size_t)H_ * I_ * 2);
  bf16_t* wh2h_b = (bf16_t*)carve((size_t)H_ * H_ * 2);
  bf16_t* wcat_b = (bf16_t*)carve((size_t)G4H_ * XHW_ * 2);
  bf16_t* wgen_b = (bf16_t*)carve((size_t)OUT_ * H_ * 2);
  bf16_t* emb_b  = (bf16_t*)carve((size_t)(V_ + 2) * E_ * 2);
  float*  bcat   = (float*) carve((size_t)G4H_ * 4);
  float*  q      = (float*) carve((size_t)B_ * H_ * 4);
  float*  gates  = (float*) carve((size_t)B_ * G4H_ * 4);
  float*  logits = (float*) carve((size_t)B_ * OUT_ * 4);
  float*  hbuf   = (float*) carve((size_t)B_ * H_ * 4);           // | contiguous
  float*  cbuf   = (float*) carve((size_t)B_ * H_ * 4);           // | zero span
  bf16_t* xh     = (bf16_t*)carve((size_t)B_ * XHW_ * 2);         // | (h,c,xh)

  auto gemm = [&](const bf16_t* A, int lda, const bf16_t* W, int ldw,
                  const float* bias, float* C, int M, int N, int K) {
    const int tiles = (M >> 4) * ((N + 15) >> 4);
    wmma_gemm_bf16<<<(tiles + 3) / 4, 128, 0, stream>>>(A, lda, W, ldw, bias, C, M, N, K);
  };
  auto gemm2x2 = [&](const bf16_t* A, int lda, const bf16_t* W, int ldw,
                     const float* bias, float* C, int M, int N, int K) {
    const int tiles = (M >> 5) * (N >> 5);
    wmma_gemm_bf16_2x2<<<(tiles + 3) / 4, 128, 0, stream>>>(A, lda, W, ldw, bias, C, M, N, K);
  };

  // ---- setup: zero state, convert to bf16, pack concatenated LSTM weight ----
  {
    const long nz = ((long)B_ * H_ * 2 + (long)B_ * XHW_ / 2);    // h + c + xh (as f32 count)
    zero_kernel<<<(unsigned)((nz + 255) / 256), 256, 0, stream>>>(hbuf, nz);
    long n;
    n = (long)B_ * S_ * I_;   cvt_kernel<<<(unsigned)((n + 255) / 256), 256, 0, stream>>>(ctx, ctx_b, n);
    n = (long)H_ * I_;        cvt_kernel<<<(unsigned)((n + 255) / 256), 256, 0, stream>>>(w_i2h, wi2h_b, n);
    n = (long)H_ * H_;        cvt_kernel<<<(unsigned)((n + 255) / 256), 256, 0, stream>>>(w_h2h, wh2h_b, n);
    n = (long)OUT_ * H_;      cvt_kernel<<<(unsigned)((n + 255) / 256), 256, 0, stream>>>(w_gen, wgen_b, n);
    n = (long)(V_ + 2) * E_;  cvt_kernel<<<(unsigned)((n + 255) / 256), 256, 0, stream>>>(emb, emb_b, n);
    n = (long)G4H_ * XHW_;    pack_cat_kernel<<<(unsigned)((n + 255) / 256), 256, 0, stream>>>(w_ih, w_hh, b_ih, b_hh, wcat_b, bcat);
  }

  // ---- one-time: h_proj = ctx @ w_i2h^T  (M=32768,N=512,K=512), 2x2 blocked ----
  gemm2x2(ctx_b, I_, wi2h_b, I_, nullptr, hproj, B_ * S_, H_, I_);

  // ---- T sequential decoder steps ----
  for (int t = 0; t < T_; ++t) {
    // q = h @ w_h2h^T + b_h2h   (A = h slice of xh, lda = XHW)
    gemm(xh + I_ + E_, XHW_, wh2h_b, H_, b_h2h, q, B_, H_, H_);
    // attention -> context (bf16) into xh[:,0:512]
    attn_kernel<<<B_, 256, 0, stream>>>(hproj, q, w_score, ctx, xh);
    // prev token embedding into xh[:,512:768]
    emb_kernel<<<B_, 256, 0, stream>>>(lbl, emb_b, xh, t);
    // gates = [ctx|emb|h] @ [w_ih|w_hh]^T + (b_ih+b_hh)  (M=128,N=2048,K=1280)
    gemm(xh, XHW_, wcat_b, XHW_, bcat, gates, B_, G4H_, XHW_);
    // LSTM cell -> h,c (f32) and h bf16 into xh[:,768:1280]
    lstm_kernel<<<B_, 256, 0, stream>>>(gates, hbuf, cbuf, xh);
    // logits = h @ w_gen^T + b_gen  (M=128,N=6001,K=512)
    gemm(xh + I_ + E_, XHW_, wgen_b, H_, b_gen, logits, B_, OUT_, H_);
    // log_softmax into output slab for step t
    lsm_kernel<<<B_, 256, 0, stream>>>(logits, out + (size_t)t * B_ * OUT_);
  }
}
